// Block_18588618457431
// MI455X (gfx1250) — matrix operand order repaired and checked by reading, not compiled, not run on hardware
//
#include <hip/hip_runtime.h>

// ============================================================================
// ViT encoder block forward on MI455X (gfx1250, wave32, WMMA).
//
// Precision strategy: all GEMMs in bf16 via v_wmma_f32_16x16x32_bf16 with f32
// accumulation; LayerNorm / softmax / GELU / residuals in f32. Weights fit in
// the 192MB L2, so a per-wave 32x64 output tile (8 WMMAs per K=32 step, A
// reuse x4 / B reuse x2) is enough; HBM traffic is ~200MB << compute time.
//
// Round-2 change: V is pre-transposed per head (LDS-tiled transpose) so the
// P@V B-fragments are two 16B row loads instead of 16 strided u16 gathers.
//
// Workspace use (~130 MB): bf16 weight copies, bf16 activations, f32 residual.
// ============================================================================

typedef __bf16 bf16;
typedef __attribute__((ext_vector_type(16))) __bf16       v16bf;
typedef __attribute__((ext_vector_type(8)))  float        v8f;
typedef __attribute__((ext_vector_type(4)))  unsigned int u32x4;
typedef __attribute__((ext_vector_type(4)))  float        f32x4;

__device__ __forceinline__ v8f wmma_bf16(v16bf a, v16bf b, v8f c) {
  // D = A(16x32) * B(32x16) + C(16x16 f32)
  return __builtin_amdgcn_wmma_f32_16x16x32_bf16(
      /*neg_a=*/false, a, /*neg_b=*/false, b,
      /*c_mod=*/(short)0, c, /*reuse_a=*/false, /*reuse_b=*/false);
}

// Build a 16-element bf16 fragment from two contiguous 8-element (16B) runs.
__device__ __forceinline__ v16bf ld_frag_pair(const bf16* p0, const bf16* p1) {
  union U { u32x4 q; bf16 e[8]; };
  U a, b;
  a.q = *reinterpret_cast<const u32x4*>(p0);
  b.q = *reinterpret_cast<const u32x4*>(p1);
  v16bf r;
#pragma unroll
  for (int i = 0; i < 8; ++i) { r[i] = a.e[i]; r[i + 8] = b.e[i]; }
  return r;
}

__device__ __forceinline__ float gelu_erf(float x) {
  return 0.5f * x * (1.0f + erff(x * 0.70710678118654752440f));
}

// ---------------------------------------------------------------------------
// fp32 -> bf16 elementwise convert (for weights)
// ---------------------------------------------------------------------------
__global__ void f32_to_bf16_kernel(const float* __restrict__ in,
                                   bf16* __restrict__ out, int n) {
  int i = blockIdx.x * blockDim.x + threadIdx.x;
  if (i < n) out[i] = (bf16)in[i];
}

// ---------------------------------------------------------------------------
// LayerNorm over C=1024, one 256-thread block per row, bf16 output
// ---------------------------------------------------------------------------
__global__ void __launch_bounds__(256)
ln_kernel(const float* __restrict__ x, const float* __restrict__ w,
          const float* __restrict__ bi, bf16* __restrict__ out) {
  const int C = 1024;
  const int row = blockIdx.x, t = threadIdx.x;
  const float* xr = x + (size_t)row * C;
  f32x4 v = *reinterpret_cast<const f32x4*>(xr + t * 4);

  __shared__ float red[256];
  red[t] = v[0] + v[1] + v[2] + v[3];
  __syncthreads();
  for (int o = 128; o > 0; o >>= 1) {
    if (t < o) red[t] += red[t + o];
    __syncthreads();
  }
  float mu = red[0] * (1.0f / 1024.0f);
  __syncthreads();

  f32x4 d = v - mu;
  red[t] = d[0] * d[0] + d[1] * d[1] + d[2] * d[2] + d[3] * d[3];
  __syncthreads();
  for (int o = 128; o > 0; o >>= 1) {
    if (t < o) red[t] += red[t + o];
    __syncthreads();
  }
  float rs = rsqrtf(red[0] * (1.0f / 1024.0f) + 1e-5f);

#pragma unroll
  for (int j = 0; j < 4; ++j) {
    int c = t * 4 + j;
    out[(size_t)row * C + c] = (bf16)(d[j] * rs * w[c] + bi[c]);
  }
}

// ---------------------------------------------------------------------------
// Generic WMMA GEMM: out[M,N] = A[M,K](bf16) * W[N,K](bf16)^T  (+bias)(+gelu)
// (+residual). One wave per 32(M) x 64(N) tile; block = 4 waves along N
// (all 4 waves share the same A rows -> L0 hits).
// ---------------------------------------------------------------------------
__global__ void __launch_bounds__(128)
gemm_wmma_kernel(const bf16* __restrict__ A, const bf16* __restrict__ W,
                 const float* __restrict__ bias, const float* __restrict__ resid,
                 float* __restrict__ outF, bf16* __restrict__ outB,
                 int M, int N, int K, int do_gelu) {
  const int lane = threadIdx.x;           // 0..31 (wave32)
  const int m0 = blockIdx.y * 32;
  const int n0 = (blockIdx.x * 4 + threadIdx.y) * 64;
  const int mlo = lane & 15, sel = lane >> 4;
  const int kb = sel * 8, kh = sel * 16;

  const v8f vzero = {0.f, 0.f, 0.f, 0.f, 0.f, 0.f, 0.f, 0.f};
  v8f acc[2][4];
#pragma unroll
  for (int t = 0; t < 2; ++t)
#pragma unroll
    for (int u = 0; u < 4; ++u) acc[t][u] = vzero;

  for (int k0 = 0; k0 < K; k0 += 32) {
    v16bf af[2], bfr[4];
#pragma unroll
    for (int t = 0; t < 2; ++t) {
      const bf16* p = A + (size_t)(m0 + t * 16 + mlo) * K + k0 + kb;
      af[t] = ld_frag_pair(p, p + 16);
    }
#pragma unroll
    for (int u = 0; u < 4; ++u) {
      const bf16* p = W + (size_t)(n0 + u * 16 + mlo) * K + k0 + kb;
      bfr[u] = ld_frag_pair(p, p + 16);
    }
#pragma unroll
    for (int t = 0; t < 2; ++t)
#pragma unroll
      for (int u = 0; u < 4; ++u)
        acc[t][u] = wmma_bf16(af[t], bfr[u], acc[t][u]);
  }

#pragma unroll
  for (int t = 0; t < 2; ++t)
#pragma unroll
    for (int u = 0; u < 4; ++u)
#pragma unroll
      for (int r = 0; r < 8; ++r) {
        const int mo = m0 + t * 16 + r + 8 * sel;
        const int no = n0 + u * 16 + mlo;
        float val = acc[t][u][r];
        if (bias) val += bias[no];
        if (do_gelu) val = gelu_erf(val);
        if (resid) val += resid[(size_t)mo * N + no];
        if (outF) outF[(size_t)mo * N + no] = val;
        if (outB) outB[(size_t)mo * N + no] = (bf16)val;
      }
}

// ---------------------------------------------------------------------------
// Per-head V transpose: qkv[(b*1024+n)*3072 + 2048 + h*64 + d]  ->
// vt[((b*16+h)*64 + d)*1024 + n].  LDS-tiled 64x64, coalesced both sides,
// +1 element padding to avoid bank conflicts.
// ---------------------------------------------------------------------------
__global__ void __launch_bounds__(256)
vtrans_kernel(const bf16* __restrict__ qkv, bf16* __restrict__ vt) {
  __shared__ bf16 tile[64 * 65];
  const int t  = threadIdx.x;
  const int kt = blockIdx.x & 15;          // 64-key tile
  const int h  = (blockIdx.x >> 4) & 15;
  const int b  = blockIdx.x >> 8;

  const bf16* src = qkv + (size_t)(b * 1024) * 3072 + 2048 + h * 64; // ld 3072
  bf16* dst = vt + (size_t)((b * 16 + h) * 64) * 1024;               // ld 1024

#pragma unroll
  for (int i = 0; i < 16; ++i) {
    int l = t + i * 256;
    int d = l & 63, kk = l >> 6;           // d fastest -> coalesced read
    tile[d * 65 + kk] = src[(size_t)(kt * 64 + kk) * 3072 + d];
  }
  __syncthreads();
#pragma unroll
  for (int i = 0; i < 16; ++i) {
    int l = t + i * 256;
    int kk = l & 63, d = l >> 6;           // kk fastest -> coalesced write
    dst[(size_t)d * 1024 + kt * 64 + kk] = tile[d * 65 + kk];
  }
}

// ---------------------------------------------------------------------------
// Attention: one wave per (b, h, 16-query tile). Scores for the full key
// range (16x1024 f32 = 64KB) live in LDS; softmax in LDS; O = P@V^T via WMMA
// with P read back from LDS as bf16 A-fragments and V^T rows as contiguous
// B-fragments. Output written back to [b*N+q, h*64+d] bf16 for proj GEMM.
// ---------------------------------------------------------------------------
__global__ void __launch_bounds__(32)
attn_kernel(const bf16* __restrict__ qkv, const bf16* __restrict__ vt,
            bf16* __restrict__ obf) {
  __shared__ float S[16 * 1024];   // 64KB of the 320KB WGP LDS

  const int lane = threadIdx.x;
  const int mlo = lane & 15, sel = lane >> 4;
  const int kb = sel * 8, kh = sel * 16;
  const int qt = blockIdx.x & 63;
  const int h  = (blockIdx.x >> 6) & 15;
  const int b  = blockIdx.x >> 10;

  const size_t ld = 3072;
  const bf16* Qb = qkv + (size_t)b * 1024 * ld + (size_t)h * 64;
  const bf16* Kb = Qb + 1024;
  const bf16* Vt = vt + (size_t)((b * 16 + h) * 64) * 1024;  // [64 x 1024]

  // Q fragments for this wave's 16 query rows (d = 0..31 and 32..63)
  const bf16* qr = Qb + (size_t)(qt * 16 + mlo) * ld;
  const v16bf qf0 = ld_frag_pair(qr + kb, qr + kb + 16);
  const v16bf qf1 = ld_frag_pair(qr + 32 + kb, qr + 32 + kb + 16);

  const v8f vzero = {0.f, 0.f, 0.f, 0.f, 0.f, 0.f, 0.f, 0.f};

  // ---- pass 1: S = (Q K^T) * 1/sqrt(64), full 16x1024 strip into LDS ----
  for (int kt = 0; kt < 64; ++kt) {
    const bf16* kr = Kb + (size_t)(kt * 16 + mlo) * ld;
    v16bf b0 = ld_frag_pair(kr + kb, kr + kb + 16);
    v16bf b1 = ld_frag_pair(kr + 32 + kb, kr + 32 + kb + 16);
    v8f c = vzero;
    c = wmma_bf16(qf0, b0, c);
    c = wmma_bf16(qf1, b1, c);
#pragma unroll
    for (int r = 0; r < 8; ++r)
      S[(r + 8 * sel) * 1024 + kt * 16 + mlo] = c[r] * 0.125f;
  }
  __syncthreads();

  // ---- softmax: 2 lanes per row (512 cols each), combine via shfl_xor ----
  float* Sr = S + mlo * 1024 + sel * 512;
  float mx = -3.4e38f;
  for (int i = 0; i < 512; ++i) mx = fmaxf(mx, Sr[i]);
  mx = fmaxf(mx, __shfl_xor(mx, 16));
  float sum = 0.0f;
  for (int i = 0; i < 512; ++i) {
    float e = __expf(Sr[i] - mx);
    Sr[i] = e;
    sum += e;
  }
  sum += __shfl_xor(sum, 16);   // lane L now holds full row-sum of row L&15
  __syncthreads();

  // ---- pass 2: O = P @ V  (K = 1024 keys, N = 64 head dims) ----
  v8f oacc[4];
#pragma unroll
  for (int u = 0; u < 4; ++u) oacc[u] = vzero;

  for (int k0 = 0; k0 < 1024; k0 += 32) {
    v16bf pa;                                  // P as A-fragment from LDS
    const float* Pr = S + mlo * 1024 + k0 + kb;
#pragma unroll
    for (int i = 0; i < 8; ++i) {
      pa[i]     = (bf16)Pr[i];
      pa[8 + i] = (bf16)Pr[16 + i];
    }
#pragma unroll
    for (int u = 0; u < 4; ++u) {
      // V^T row (head-dim) is the B-matrix column -> two 16B loads (k = kb.., 16+kb..)
      const bf16* p = Vt + (size_t)(u * 16 + mlo) * 1024 + k0 + kb;
      oacc[u] = wmma_bf16(pa, ld_frag_pair(p, p + 16), oacc[u]);
    }
  }

#pragma unroll
  for (int u = 0; u < 4; ++u)
#pragma unroll
    for (int r = 0; r < 8; ++r) {
      const int mo = r + 8 * sel;
      const float denom = __shfl(sum, mo);
      const float val = oacc[u][r] / denom;
      obf[(size_t)(b * 1024 + qt * 16 + mo) * 1024 + h * 64 + u * 16 + mlo] =
          (bf16)val;
    }
}

// ---------------------------------------------------------------------------
// Host-side orchestration
// ---------------------------------------------------------------------------
extern "C" void kernel_launch(void* const* d_in, const int* in_sizes, int n_in,
                              void* d_out, int out_size, void* d_ws, size_t ws_size,
                              hipStream_t stream) {
  (void)in_sizes; (void)n_in; (void)out_size; (void)ws_size;

  const float* x      = (const float*)d_in[0];
  const float* ln1_w  = (const float*)d_in[1];
  const float* ln1_b  = (const float*)d_in[2];
  const float* qkv_w  = (const float*)d_in[3];
  const float* proj_w = (const float*)d_in[4];
  const float* proj_b = (const float*)d_in[5];
  const float* ln2_w  = (const float*)d_in[6];
  const float* ln2_b  = (const float*)d_in[7];
  const float* fc1_w  = (const float*)d_in[8];
  const float* fc1_b  = (const float*)d_in[9];
  const float* fc2_w  = (const float*)d_in[10];
  const float* fc2_b  = (const float*)d_in[11];
  float* out = (float*)d_out;

  const int M = 4096;        // B*N rows
  const int C = 1024;
  const int H3 = 3072;       // 3C
  const int HID = 4096;      // 4C

  // Carve workspace (~130 MB total)
  char* p = (char*)d_ws;
  auto carve = [&](size_t bytes) {
    char* r = p;
    p += (bytes + 255) & ~(size_t)255;
    return r;
  };
  bf16* w_qkv  = (bf16*)carve((size_t)H3 * C * 2);        // 6 MB
  bf16* w_proj = (bf16*)carve((size_t)C * C * 2);         // 2 MB
  bf16* w_fc1  = (bf16*)carve((size_t)HID * C * 2);       // 8 MB
  bf16* w_fc2  = (bf16*)carve((size_t)C * HID * 2);       // 8 MB
  bf16* h_bf   = (bf16*)carve((size_t)M * C * 2);         // 8 MB  (ln1 out)
  bf16* qkv_bf = (bf16*)carve((size_t)M * H3 * 2);        // 24 MB
  bf16* vt_bf  = (bf16*)carve((size_t)M * C * 2);         // 8 MB  (V^T per head)
  bf16* o_bf   = (bf16*)carve((size_t)M * C * 2);         // 8 MB  (attn out)
  float* x1    = (float*)carve((size_t)M * C * 4);        // 16 MB (residual 1)
  bf16* h2_bf  = (bf16*)carve((size_t)M * C * 2);         // 8 MB  (ln2 out)
  bf16* h3_bf  = (bf16*)carve((size_t)M * HID * 2);       // 32 MB (gelu out)

  // 1) weights -> bf16
  auto cvt = [&](const float* src, bf16* dst, int n) {
    f32_to_bf16_kernel<<<dim3((n + 255) / 256), dim3(256), 0, stream>>>(src, dst, n);
  };
  cvt(qkv_w,  w_qkv,  H3 * C);
  cvt(proj_w, w_proj, C * C);
  cvt(fc1_w,  w_fc1,  HID * C);
  cvt(fc2_w,  w_fc2,  C * HID);

  // 2) LN1
  ln_kernel<<<dim3(M), dim3(256), 0, stream>>>(x, ln1_w, ln1_b, h_bf);

  // 3) QKV = h @ qkv_w^T (no bias), bf16 out
  gemm_wmma_kernel<<<dim3(H3 / 256, M / 32), dim3(32, 4), 0, stream>>>(
      h_bf, w_qkv, nullptr, nullptr, nullptr, qkv_bf, M, H3, C, 0);

  // 3b) V -> V^T per head (4*16*16 = 1024 tiles of 64x64)
  vtrans_kernel<<<dim3(1024), dim3(256), 0, stream>>>(qkv_bf, vt_bf);

  // 4) attention: 4 batches * 16 heads * 64 query tiles = 4096 waves
  attn_kernel<<<dim3(4096), dim3(32), 0, stream>>>(qkv_bf, vt_bf, o_bf);

  // 5) x1 = x + o @ proj_w^T + proj_b
  gemm_wmma_kernel<<<dim3(C / 256, M / 32), dim3(32, 4), 0, stream>>>(
      o_bf, w_proj, proj_b, x, x1, nullptr, M, C, C, 0);

  // 6) LN2
  ln_kernel<<<dim3(M), dim3(256), 0, stream>>>(x1, ln2_w, ln2_b, h2_bf);

  // 7) h3 = gelu(h2 @ fc1_w^T + fc1_b), bf16 out
  gemm_wmma_kernel<<<dim3(HID / 256, M / 32), dim3(32, 4), 0, stream>>>(
      h2_bf, w_fc1, fc1_b, nullptr, nullptr, h3_bf, M, HID, C, 1);

  // 8) out = x1 + h3 @ fc2_w^T + fc2_b
  gemm_wmma_kernel<<<dim3(C / 256, M / 32), dim3(32, 4), 0, stream>>>(
      h3_bf, w_fc2, fc2_b, x1, out, nullptr, M, C, HID, 0);
}
